// Model_30331059045134
// MI455X (gfx1250) — matrix operand order repaired and checked by reading, not compiled, not run on hardware
//
#include <hip/hip_runtime.h>
#include <stdint.h>

// ---------------- problem constants ----------------
#define N_    64
#define CIN   64
#define T_    256
#define V_    17
#define OUTC  128
#define KSUB  3
#define KC    384          // conv output rows (3*128)
#define COLS  512          // conv (384) + down (128) GEMM columns
#define TV    4352         // T*V
#define TTILE 8
#define PTILE 136          // TTILE*V
#define NTT   32           // T/TTILE
#define APAD  72           // A-stage row stride in bf16 (64 ci + pad, 144B rows)
#define MROW  136          // m_lds row stride in bf16
#define EPS_  1e-5f

// K3 LDS map: m_lds [0,104448) | bf16 A-stage [104448,125184) | f32 DMA stage [125184,160000)
//             A_final overlay  [104448,159936) in phase 2
#define K3_LDS_ASTAGE 104448
#define K3_LDS_FSTAGE 125184
#define K3_LDS_BYTES  160000
// K5 LDS map: bf16 A-stage [0,20736) | f32 DMA stage [20736,55552)
#define K5_LDS_FSTAGE 20736
#define K5_LDS_BYTES  55552

typedef __attribute__((ext_vector_type(16))) __bf16 v16bf;
typedef __attribute__((ext_vector_type(8)))  __bf16 v8bf;
typedef __attribute__((ext_vector_type(8)))  float  v8f;
typedef __attribute__((ext_vector_type(4)))  unsigned int v4u;
typedef __attribute__((ext_vector_type(8)))  int v8i;
typedef __attribute__((ext_vector_type(4)))  int v4i;

__device__ __forceinline__ v8f wmma_bf16(v16bf a, v16bf b, v8f c) {
  return __builtin_amdgcn_wmma_f32_16x16x32_bf16(false, a, false, b, (short)0, c, false, false);
}

// ---- Tensor Data Mover: 2D tile load x0[64 ci][136 pos] f32 (row stride TV) -> LDS ----
#if __has_builtin(__builtin_amdgcn_tensor_load_to_lds)
#define HAVE_TDM 1
__device__ __forceinline__ void tdm_load_x0_tile(const float* gptr, unsigned lds_off) {
  unsigned long long ga = (unsigned long long)(uintptr_t)gptr;
  v4u g0;
  g0[0] = 1u;                                               // count=1 (valid user D#)
  g0[1] = lds_off;                                          // lds_addr (bytes)
  g0[2] = (unsigned)ga;                                     // global_addr[31:0]
  g0[3] = (unsigned)((ga >> 32) & 0x01FFFFFFu) | (2u << 30);// global_addr[56:32] | type=2
  v8i g1;
  g1[0] = 0x00020000;                                       // wg_mask=0 | data_size=2 (4B)
  g1[1] = (int)((TV & 0xFFFF) << 16);                       // tensor_dim0[15:0] @bits63:48
  g1[2] = (int)((TV >> 16) | (CIN << 16));                  // dim0[31:16] | tensor_dim1[15:0]
  g1[3] = (int)(PTILE << 16);                               // dim1[31:16]=0 | tile_dim0=136
  g1[4] = CIN;                                              // tile_dim1=64 | tile_dim2=0
  g1[5] = TV;                                               // tensor_dim0_stride[31:0]
  g1[6] = 0; g1[7] = 0;
  v4i z = {0, 0, 0, 0};
#if defined(__clang_major__) && (__clang_major__ >= 23)
  v8i z8 = {0, 0, 0, 0, 0, 0, 0, 0};
  __builtin_amdgcn_tensor_load_to_lds(g0, g1, z, z, z8, 0);
#else
  __builtin_amdgcn_tensor_load_to_lds(g0, g1, z, z, 0);
#endif
}
#endif

// ---------------- workspace layout (bytes) ----------------
constexpr size_t OFF_MOUT = 0;                     // 64*128*4352 f32 = 142606336
constexpr size_t OFF_AFIN = 142606336;             // 64*96*289 bf16  = 3551232
constexpr size_t OFF_WB   = 146157568;             // 512*64 bf16     = 65536
constexpr size_t OFF_BIAS = 146223104;             // 512 f32
constexpr size_t OFF_XBAR = 146225152;             // 64*64*17 f32
constexpr size_t OFF_S1   = 146503680;             // 64*128 f32
constexpr size_t OFF_S2   = 146536448;             // 64*128 f32
constexpr size_t OFF_DSUM = 146569216;             // 128 f32
constexpr size_t OFF_DSS  = 146569728;             // 128 f32
constexpr size_t OFF_GATE = 146570240;             // 64*128 f32
constexpr size_t OFF_COEF = 146603008;             // 4*128 f32 (sM,tM,sD,tD)

// ================= K0: pack bf16 weights, zero accumulators =================
__global__ void k0_prep(const float* __restrict__ conv_w, const float* __restrict__ conv_b,
                        const float* __restrict__ down_w, const float* __restrict__ down_b,
                        __bf16* __restrict__ Wb, float* __restrict__ bias,
                        float* __restrict__ S1, float* __restrict__ S2,
                        float* __restrict__ dsum, float* __restrict__ dssum) {
  int idx = blockIdx.x * 256 + threadIdx.x;
  if (idx < COLS * CIN) {
    int col = idx >> 6, ci = idx & 63;
    float w = (col < KC) ? conv_w[col * CIN + ci] : down_w[(col - KC) * CIN + ci];
    Wb[col * CIN + ci] = (__bf16)w;                 // [col][ci], B is K(ci) x N(col)
  }
  if (idx < COLS) bias[idx] = (idx < KC) ? conv_b[idx] : down_b[idx - KC];
  if (idx < N_ * OUTC) { S1[idx] = 0.f; S2[idx] = 0.f; }
  if (idx < OUTC) { dsum[idx] = 0.f; dssum[idx] = 0.f; }
}

// ================= K1: xbar[n,ci,v] = mean_t x0 =================
__global__ void k1_xbar(const float* __restrict__ x0, float* __restrict__ xbar) {
  int ci = blockIdx.x, n = blockIdx.y, t = threadIdx.x;      // 256 threads = T
  __shared__ float red[256][18];
  const float* row = x0 + ((size_t)(n * CIN + ci)) * TV + t * V_;
#pragma unroll
  for (int v = 0; v < V_; v++) red[t][v] = row[v];
  __syncthreads();
  for (int s = 128; s > 0; s >>= 1) {
    if (t < s) {
#pragma unroll
      for (int v = 0; v < V_; v++) red[t][v] += red[t + s][v];
    }
    __syncthreads();
  }
  if (t < V_) xbar[(n * CIN + ci) * V_ + t] = red[0][t] * (1.f / (float)T_);
}

// ================= K2: attention -> A_final (bf16), one block per n ========
__global__ void k2_atten(const float* __restrict__ xbar,
                         const float* __restrict__ conv_w, const float* __restrict__ conv_b,
                         const float* __restrict__ W_spa, const float* __restrict__ b_spa,
                         const float* __restrict__ W_mix, const float* __restrict__ b_mix,
                         const float* __restrict__ A_GEME, const float* __restrict__ A_SE,
                         __bf16* __restrict__ Afin) {
  int n = blockIdx.x, tid = threadIdx.x;
  __shared__ float xb[CIN][V_];
  __shared__ float Wq[12][CIN], Wk[12][CIN];
  __shared__ float bq[12], bk[12];
  __shared__ float Q0[12][V_], K0s[12][V_];
  __shared__ float Qs[12][V_], Ks[12][V_], Qm[12][V_], Km[12][V_];
  __shared__ float S[12][V_][V_];
  __shared__ float cross[3][3][V_][V_];

  for (int i = tid; i < CIN * V_; i += 256) {
    int ci = i / V_, v = i - ci * V_;
    xb[ci][v] = xbar[(n * CIN + ci) * V_ + v];
  }
  for (int i = tid; i < 12 * CIN; i += 256) {     // pooled conv weights (Q: j<16, K: j>=16)
    int kq = i >> 6, ci = i & 63;
    int ks = kq >> 2, qt = kq & 3;
    const float* base = conv_w + (size_t)(ks * OUTC + qt * 32) * CIN + ci;
    float sq = 0.f, sk = 0.f;
    for (int j = 0; j < 16; j++) { sq += base[j * CIN]; sk += base[(j + 16) * CIN]; }
    Wq[kq][ci] = sq * (1.f / 16.f); Wk[kq][ci] = sk * (1.f / 16.f);
  }
  if (tid < 12) {
    int ks = tid >> 2, qt = tid & 3;
    const float* bb = conv_b + ks * OUTC + qt * 32;
    float sq = 0.f, sk = 0.f;
    for (int j = 0; j < 16; j++) { sq += bb[j]; sk += bb[j + 16]; }
    bq[tid] = sq * (1.f / 16.f); bk[tid] = sk * (1.f / 16.f);
  }
  __syncthreads();
  for (int i = tid; i < 12 * V_; i += 256) {      // pooled Q/K (linearity of pooling)
    int kq = i / V_, v = i - kq * V_;
    float aq = bq[kq], ak = bk[kq];
    for (int ci = 0; ci < CIN; ci++) { aq += Wq[kq][ci] * xb[ci][v]; ak += Wk[kq][ci] * xb[ci][v]; }
    Q0[kq][v] = aq; K0s[kq][v] = ak;
  }
  __syncthreads();
  for (int i = tid; i < 12 * V_; i += 256) {      // linear layers + relu
    int kq = i / V_, w = i - kq * V_;
    float qs = b_spa[w], ks_ = b_spa[w], qm = b_mix[w], km = b_mix[w];
#pragma unroll
    for (int v = 0; v < V_; v++) {
      float ws_ = W_spa[w * V_ + v], wm = W_mix[w * V_ + v];
      qs += Q0[kq][v] * ws_; ks_ += K0s[kq][v] * ws_;
      qm += Q0[kq][v] * wm;  km += K0s[kq][v] * wm;
    }
    Qs[kq][w] = fmaxf(qs, 0.f); Ks[kq][w] = fmaxf(ks_, 0.f);
    Qm[kq][w] = fmaxf(qm, 0.f); Km[kq][w] = fmaxf(km, 0.f);
  }
  __syncthreads();
  for (int i = tid; i < 12 * V_; i += 256) {      // S = softmax_j(Qs_i * Ks_j)
    int kq = i / V_, ii = i - kq * V_;
    float e[V_]; float mx = -1e30f;
#pragma unroll
    for (int j = 0; j < V_; j++) { e[j] = Qs[kq][ii] * Ks[kq][j]; mx = fmaxf(mx, e[j]); }
    float sm = 0.f;
#pragma unroll
    for (int j = 0; j < V_; j++) { e[j] = __expf(e[j] - mx); sm += e[j]; }
    float inv = 1.f / sm;
#pragma unroll
    for (int j = 0; j < V_; j++) S[kq][ii][j] = e[j] * inv;
  }
  for (int i = tid; i < 9 * V_; i += 256) {       // cross = softmax_j(Km[r]_i * Qm[r+1]_j)
    int ksr = i / V_, ii = i - ksr * V_;
    int ks = ksr / 3, r = ksr - ks * 3;
    float km = Km[ks * 4 + r][ii];
    float e[V_]; float mx = -1e30f;
#pragma unroll
    for (int j = 0; j < V_; j++) { e[j] = km * Qm[ks * 4 + r + 1][j]; mx = fmaxf(mx, e[j]); }
    float sm = 0.f;
#pragma unroll
    for (int j = 0; j < V_; j++) { e[j] = __expf(e[j] - mx); sm += e[j]; }
    float inv = 1.f / sm;
#pragma unroll
    for (int j = 0; j < V_; j++) cross[ks][r][ii][j] = e[j] * inv;
  }
  __syncthreads();
  // A_final[n,ks,qt,g] = 0.5*atten + (A_SE+A_GEME)[ks,g]; atten = S/2 + cross/4 combos
  for (int i = tid; i < KSUB * 4 * 8 * V_; i += 256) {
    int v = i % V_; int i2 = i / V_;
    int g = i2 & 7; i2 >>= 3;
    int qt = i2 & 3; int ks = i2 >> 2;
    int kq = ks * 4 + qt;
    const float* ase = A_SE   + ((size_t)(ks * 8 + g) * V_ + v) * V_;
    const float* age = A_GEME + ((size_t)(ks * 8 + g) * V_ + v) * V_;
    __bf16* dst = Afin + (((size_t)n * KSUB + ks) * 32 + qt * 8 + g) * 289 + v * V_;
#pragma unroll
    for (int w = 0; w < V_; w++) {
      float at = 0.5f * S[kq][v][w];
      if (qt == 0)      at += 0.25f * cross[ks][0][v][w];
      else if (qt == 1) at += 0.25f * (cross[ks][0][v][w] + cross[ks][1][v][w]);
      else if (qt == 2) at += 0.25f * (cross[ks][1][v][w] + cross[ks][2][v][w]);
      else              at += 0.25f * cross[ks][2][v][w];
      dst[w] = (__bf16)(0.5f * at + ase[w] + age[w]);
    }
  }
}

// ================= K3: fused conv-GEMM (WMMA bf16) + graph conv =================
// block = (tt, n); 256 threads = 8 wave32. Waves 0-5: conv cols, waves 6-7: down cols.
__global__ void __launch_bounds__(256)
k3_main(const float* __restrict__ x0, const __bf16* __restrict__ Wb,
        const float* __restrict__ bias, const __bf16* __restrict__ Afin,
        float* __restrict__ m_out, float* __restrict__ S1, float* __restrict__ S2,
        float* __restrict__ dsum, float* __restrict__ dssum) {
  extern __shared__ char smem[];
  __bf16* m_lds  = (__bf16*)smem;                        // 384*136
  __bf16* astage = (__bf16*)(smem + K3_LDS_ASTAGE);      // 144 rows x 72 (phase 1)
  __bf16* afin_l = (__bf16*)(smem + K3_LDS_ASTAGE);      // 96*289      (phase 2 overlay)
  const int tt = blockIdx.x, n = blockIdx.y, tid = threadIdx.x;

  // ---- phase 1a: stage x0 tile -> bf16 [p][ci], zero-pad p=136..143 ----
  const float* xbp = x0 + (size_t)n * CIN * TV + tt * PTILE;
#ifdef HAVE_TDM
  if (tid < 32) {                                        // wave 0 drives the TDM
    tdm_load_x0_tile(xbp, K3_LDS_FSTAGE);
    __builtin_amdgcn_s_wait_tensorcnt(0);
  }
  __syncthreads();
  const float* fst = (const float*)(smem + K3_LDS_FSTAGE);
  for (int i = tid; i < CIN * PTILE; i += 256) {         // f32 LDS -> bf16 transposed
    int ci = i / PTILE, p = i - ci * PTILE;
    astage[p * APAD + ci] = (__bf16)fst[i];
  }
#else
  for (int i = tid; i < CIN * PTILE; i += 256) {
    int ci = i / PTILE, p = i - ci * PTILE;
    astage[p * APAD + ci] = (__bf16)xbp[(size_t)ci * TV + p];
  }
#endif
  for (int i = tid; i < 8 * CIN; i += 256) {
    int p = 136 + (i >> 6), ci = i & 63;
    astage[p * APAD + ci] = (__bf16)0.f;
  }
  if (tid < CIN && tt + 1 < NTT)                          // gfx1250 global_prefetch_b8
    __builtin_prefetch(xbp + (size_t)tid * TV + PTILE, 0, 1);
  __syncthreads();

  // ---- phase 1b: GEMM  D[pos, col] = A[pos, ci] * W[ci, col], K=64 (2 WMMA steps) ----
  const int wave = tid >> 5, lane = tid & 31;
  const int lhalf = lane >> 4, l15 = lane & 15;
  const int nt0 = wave * 4;                                // 4 N-tiles per wave, 32 total
  const int wuni = __builtin_amdgcn_readfirstlane(wave);   // scalar wave id
  v16bf bfrag[4][2];
  float bias_r[4];
#pragma unroll
  for (int j = 0; j < 4; j++) {
    int col = (nt0 + j) * 16 + l15;                        // B: lane=N, elem e -> K=16*lhalf+e
    bfrag[j][0] = *(const v16bf*)(Wb + col * CIN + 16 * lhalf);
    bfrag[j][1] = *(const v16bf*)(Wb + col * CIN + 32 + 16 * lhalf);
    bias_r[j] = bias[col];
  }
  auto load_a = [&](int mt, v16bf& a0, v16bf& a1) {
    const __bf16* abase = astage + (mt * 16 + l15) * APAD; // A: lane=M
    v8bf a0lo = *(const v8bf*)(abase + 16 * lhalf);        // K 0-7 / 16-23
    v8bf a0hi = *(const v8bf*)(abase + 8 + 16 * lhalf);    // K 8-15 / 24-31
    v8bf a1lo = *(const v8bf*)(abase + 32 + 16 * lhalf);   // K 32-39 / 48-55
    v8bf a1hi = *(const v8bf*)(abase + 40 + 16 * lhalf);   // K 40-47 / 56-63
#pragma unroll
    for (int e = 0; e < 8; e++) {
      a0[e] = a0lo[e]; a0[e + 8] = a0hi[e];
      a1[e] = a1lo[e]; a1[e + 8] = a1hi[e];
    }
  };
  if (wuni < 6) {                                          // conv waves -> m tile in LDS
#pragma unroll 1
    for (int mt = 0; mt < 8; mt++) {                       // steady state: no guards
      v16bf a0, a1; load_a(mt, a0, a1);
      const int pbase = mt * 16 + 8 * lhalf;
#pragma unroll
      for (int j = 0; j < 4; j++) {
        v8f acc = {};
        acc = wmma_bf16(a0, bfrag[j][0], acc);
        acc = wmma_bf16(a1, bfrag[j][1], acc);
        int col = (nt0 + j) * 16 + l15;
        v8bf st;
#pragma unroll
        for (int r = 0; r < 8; r++) st[r] = (__bf16)(acc[r] + bias_r[j]);
        *(v8bf*)(m_lds + col * MROW + pbase) = st;
      }
    }
    { // peeled mt=8: only lower half-rows (p=128..135) are real
      v16bf a0, a1; load_a(8, a0, a1);
#pragma unroll
      for (int j = 0; j < 4; j++) {
        v8f acc = {};
        acc = wmma_bf16(a0, bfrag[j][0], acc);
        acc = wmma_bf16(a1, bfrag[j][1], acc);
        int col = (nt0 + j) * 16 + l15;
        if (lhalf == 0) {
          v8bf st;
#pragma unroll
          for (int r = 0; r < 8; r++) st[r] = (__bf16)(acc[r] + bias_r[j]);
          *(v8bf*)(m_lds + col * MROW + 128) = st;
        }
      }
    }
  } else {                                                 // down waves -> BN stats only
    float ds1[4] = {0.f, 0.f, 0.f, 0.f}, ds2[4] = {0.f, 0.f, 0.f, 0.f};
#pragma unroll 1
    for (int mt = 0; mt < 8; mt++) {
      v16bf a0, a1; load_a(mt, a0, a1);
#pragma unroll
      for (int j = 0; j < 4; j++) {
        v8f acc = {};
        acc = wmma_bf16(a0, bfrag[j][0], acc);
        acc = wmma_bf16(a1, bfrag[j][1], acc);
#pragma unroll
        for (int r = 0; r < 8; r++) {
          float d = acc[r] + bias_r[j];
          ds1[j] += d; ds2[j] += d * d;
        }
      }
    }
    { // peeled mt=8
      v16bf a0, a1; load_a(8, a0, a1);
#pragma unroll
      for (int j = 0; j < 4; j++) {
        v8f acc = {};
        acc = wmma_bf16(a0, bfrag[j][0], acc);
        acc = wmma_bf16(a1, bfrag[j][1], acc);
        if (lhalf == 0) {
#pragma unroll
          for (int r = 0; r < 8; r++) {
            float d = acc[r] + bias_r[j];
            ds1[j] += d; ds2[j] += d * d;
          }
        }
      }
    }
#pragma unroll
    for (int j = 0; j < 4; j++) {                          // lane^16 holds same column
      int col = (nt0 + j) * 16 + l15;
      float o1 = __shfl_xor(ds1[j], 16, 32);
      float o2 = __shfl_xor(ds2[j], 16, 32);
      if (lhalf == 0) {
        atomicAdd(&dsum[col - KC], ds1[j] + o1);
        atomicAdd(&dssum[col - KC], ds2[j] + o2);
      }
    }
  }
  __syncthreads();

  // ---- phase 2a: pull this sample's 96 A_final matrices into LDS (over A-stage) ----
  {
    const uint32_t* src = (const uint32_t*)(Afin + (size_t)n * 96 * 289);
    uint32_t* dst = (uint32_t*)afin_l;
    for (int i = tid; i < (96 * 289 * 2) / 4; i += 256) dst[i] = src[i];
  }
  __syncthreads();

  // ---- phase 2b: graph conv  m_out[c,t,w] = sum_ks sum_v m[ks,c,t,v]*Afin[ks,qg(c),v,w]
  const int c = tid & 127;
  const int tstart = tid >> 7;                              // thread covers t = tstart+2a
  const int qg = ((c >> 5) << 3) | (c & 7);                 // quarter*8 + group
  float accw[4][V_];
#pragma unroll
  for (int a = 0; a < 4; a++)
#pragma unroll
    for (int w = 0; w < V_; w++) accw[a][w] = 0.f;
#pragma unroll
  for (int ks = 0; ks < KSUB; ks++) {
    const __bf16* arow = afin_l + (ks * 32 + qg) * 289;
    const __bf16* mrow = m_lds + (ks * OUTC + c) * MROW;
#pragma unroll
    for (int v = 0; v < V_; v++) {
      float mv[4];
#pragma unroll
      for (int a = 0; a < 4; a++) mv[a] = (float)mrow[(tstart + 2 * a) * V_ + v];
#pragma unroll
      for (int w = 0; w < V_; w++) {
        float av = (float)arow[v * V_ + w];
#pragma unroll
        for (int a = 0; a < 4; a++) accw[a][w] += mv[a] * av;
      }
    }
  }
  float s1 = 0.f, s2 = 0.f;
  float* mo = m_out + ((size_t)(n * OUTC + c)) * TV + tt * PTILE;
#pragma unroll
  for (int a = 0; a < 4; a++) {
    int t = tstart + 2 * a;
#pragma unroll
    for (int w = 0; w < V_; w++) {
      float vv = accw[a][w];
      mo[t * V_ + w] = vv;
      s1 += vv; s2 += vv * vv;
    }
  }
  atomicAdd(&S1[n * OUTC + c], s1);                         // per-(n,c) sums for gate + BN
  atomicAdd(&S2[n * OUTC + c], s2);
}

// ================= K4: gate + BN affine coefficients (1 block, 128 threads) =====
__global__ void k4_coeffs(const float* __restrict__ S1, const float* __restrict__ S2,
                          const float* __restrict__ dsum, const float* __restrict__ dssum,
                          const float* __restrict__ charef_w,
                          const float* __restrict__ bn_gamma, const float* __restrict__ bn_beta,
                          const float* __restrict__ dbn_gamma, const float* __restrict__ dbn_beta,
                          float* __restrict__ gate, float* __restrict__ coef) {
  __shared__ float s1s[N_][OUTC];
  int tid = threadIdx.x;
  for (int i = tid; i < N_ * OUTC; i += 128) s1s[i >> 7][i & 127] = S1[i];
  __syncthreads();
  const int c = tid;
  const float w0 = charef_w[0], w1 = charef_w[1], w2 = charef_w[2];
  const float invTV = 1.f / (float)TV;
  float sumM = 0.f, sqM = 0.f;
  for (int n = 0; n < N_; n++) {
    float crm = (c > 0)   ? s1s[n][c - 1] * invTV : 0.f;    // zero-padded channel conv
    float cr0 = s1s[n][c] * invTV;
    float crp = (c < 127) ? s1s[n][c + 1] * invTV : 0.f;
    float g = 1.f / (1.f + __expf(-(w0 * crm + w1 * cr0 + w2 * crp)));
    gate[n * OUTC + c] = g;
    float f = 1.f + g;                                      // CR_out = m_out*(1+gate)
    sumM += f * s1s[n][c];
    sqM  += f * f * S2[n * OUTC + c];
  }
  const float invNTV = 1.f / ((float)N_ * (float)TV);
  float meanM = sumM * invNTV;
  float varM = sqM * invNTV - meanM * meanM;
  float sM = bn_gamma[c] * rsqrtf(varM + EPS_);
  float tM = bn_beta[c] - meanM * sM;
  float meanD = dsum[c] * invNTV;
  float varD = dssum[c] * invNTV - meanD * meanD;
  float sD = dbn_gamma[c] * rsqrtf(varD + EPS_);
  float tD = dbn_beta[c] - meanD * sD;
  coef[c] = sM; coef[OUTC + c] = tM; coef[2 * OUTC + c] = sD; coef[3 * OUTC + c] = tD;
}

// ================= K5: down-GEMM (WMMA) + gate + both BNs + residual + relu =====
__global__ void __launch_bounds__(256)
k5_final(const float* __restrict__ x0, const __bf16* __restrict__ Wb,
         const float* __restrict__ bias, const float* __restrict__ m_out,
         const float* __restrict__ gate, const float* __restrict__ coef,
         float* __restrict__ out) {
  extern __shared__ char smem[];
  __bf16* astage = (__bf16*)smem;                          // 144 x 72 bf16
  const int tt = blockIdx.x, n = blockIdx.y, tid = threadIdx.x;
  const float* xbp = x0 + (size_t)n * CIN * TV + tt * PTILE;
#ifdef HAVE_TDM
  if (tid < 32) {
    tdm_load_x0_tile(xbp, K5_LDS_FSTAGE);
    __builtin_amdgcn_s_wait_tensorcnt(0);
  }
  __syncthreads();
  const float* fst = (const float*)(smem + K5_LDS_FSTAGE);
  for (int i = tid; i < CIN * PTILE; i += 256) {
    int ci = i / PTILE, p = i - ci * PTILE;
    astage[p * APAD + ci] = (__bf16)fst[i];
  }
#else
  for (int i = tid; i < CIN * PTILE; i += 256) {
    int ci = i / PTILE, p = i - ci * PTILE;
    astage[p * APAD + ci] = (__bf16)xbp[(size_t)ci * TV + p];
  }
#endif
  for (int i = tid; i < 8 * CIN; i += 256) {
    int p = 136 + (i >> 6), ci = i & 63;
    astage[p * APAD + ci] = (__bf16)0.f;
  }
  __syncthreads();
  const int lane = tid & 31;
  const int lhalf = lane >> 4, l15 = lane & 15;
  const int col = KC + (tid >> 5) * 16 + l15;              // down columns 384..511
  const int c = col - KC;
  v16bf b0 = *(const v16bf*)(Wb + col * CIN + 16 * lhalf);
  v16bf b1 = *(const v16bf*)(Wb + col * CIN + 32 + 16 * lhalf);
  const float bi = bias[col];
  const float sM = coef[c], tM = coef[OUTC + c], sD = coef[2 * OUTC + c], tD = coef[3 * OUTC + c];
  const float f = 1.f + gate[n * OUTC + c];
  const float* mo = m_out + ((size_t)(n * OUTC + c)) * TV + tt * PTILE;
  float* oo = out + ((size_t)(n * OUTC + c)) * TV + tt * PTILE;
  auto load_a = [&](int mt, v16bf& a0, v16bf& a1) {
    const __bf16* abase = astage + (mt * 16 + l15) * APAD;
    v8bf a0lo = *(const v8bf*)(abase + 16 * lhalf);
    v8bf a0hi = *(const v8bf*)(abase + 8 + 16 * lhalf);
    v8bf a1lo = *(const v8bf*)(abase + 32 + 16 * lhalf);
    v8bf a1hi = *(const v8bf*)(abase + 40 + 16 * lhalf);
#pragma unroll
    for (int e = 0; e < 8; e++) {
      a0[e] = a0lo[e]; a0[e + 8] = a0hi[e];
      a1[e] = a1lo[e]; a1[e + 8] = a1hi[e];
    }
  };
#pragma unroll 1
  for (int mt = 0; mt < 8; mt++) {                          // steady state: no guards
    v16bf a0, a1; load_a(mt, a0, a1);
    v8f acc = {};
    acc = wmma_bf16(a0, b0, acc);
    acc = wmma_bf16(a1, b1, acc);
    const int pbase = mt * 16 + 8 * lhalf;
#pragma unroll
    for (int r = 0; r < 8; r++) {
      float d = acc[r] + bi;                                // down + bias
      float y = mo[pbase + r] * f * sM + tM + d * sD + tD;
      oo[pbase + r] = fmaxf(y, 0.f);
    }
  }
  { // peeled mt=8: only p=128..135 are real rows
    v16bf a0, a1; load_a(8, a0, a1);
    v8f acc = {};
    acc = wmma_bf16(a0, b0, acc);
    acc = wmma_bf16(a1, b1, acc);
    if (lhalf == 0) {
#pragma unroll
      for (int r = 0; r < 8; r++) {
        float d = acc[r] + bi;
        float y = mo[128 + r] * f * sM + tM + d * sD + tD;
        oo[128 + r] = fmaxf(y, 0.f);
      }
    }
  }
}

// ======================= launcher =======================
extern "C" void kernel_launch(void* const* d_in, const int* in_sizes, int n_in,
                              void* d_out, int out_size, void* d_ws, size_t ws_size,
                              hipStream_t stream) {
  const float* x0        = (const float*)d_in[0];
  const float* conv_w    = (const float*)d_in[1];
  const float* conv_b    = (const float*)d_in[2];
  const float* W_spa     = (const float*)d_in[3];
  const float* b_spa     = (const float*)d_in[4];
  const float* W_mix     = (const float*)d_in[5];
  const float* b_mix     = (const float*)d_in[6];
  const float* A_GEME    = (const float*)d_in[7];
  const float* A_SE      = (const float*)d_in[8];
  const float* charef_w  = (const float*)d_in[9];
  const float* bn_gamma  = (const float*)d_in[10];
  const float* bn_beta   = (const float*)d_in[11];
  const float* down_w    = (const float*)d_in[12];
  const float* down_b    = (const float*)d_in[13];
  const float* dbn_gamma = (const float*)d_in[14];
  const float* dbn_beta  = (const float*)d_in[15];
  (void)in_sizes; (void)n_in; (void)out_size; (void)ws_size;

  char* ws = (char*)d_ws;
  float*  m_out = (float*)(ws + OFF_MOUT);
  __bf16* Afin  = (__bf16*)(ws + OFF_AFIN);
  __bf16* Wb    = (__bf16*)(ws + OFF_WB);
  float*  biasv = (float*)(ws + OFF_BIAS);
  float*  xbar  = (float*)(ws + OFF_XBAR);
  float*  S1    = (float*)(ws + OFF_S1);
  float*  S2    = (float*)(ws + OFF_S2);
  float*  dsum  = (float*)(ws + OFF_DSUM);
  float*  dssum = (float*)(ws + OFF_DSS);
  float*  gatep = (float*)(ws + OFF_GATE);
  float*  coef  = (float*)(ws + OFF_COEF);
  float*  outp  = (float*)d_out;

  k0_prep<<<128, 256, 0, stream>>>(conv_w, conv_b, down_w, down_b, Wb, biasv, S1, S2, dsum, dssum);
  k1_xbar<<<dim3(CIN, N_), 256, 0, stream>>>(x0, xbar);
  k2_atten<<<N_, 256, 0, stream>>>(xbar, conv_w, conv_b, W_spa, b_spa, W_mix, b_mix,
                                   A_GEME, A_SE, Afin);
  k3_main<<<dim3(NTT, N_), 256, K3_LDS_BYTES, stream>>>(
      x0, Wb, biasv, Afin, m_out, S1, S2, dsum, dssum);
  k4_coeffs<<<1, 128, 0, stream>>>(S1, S2, dsum, dssum, charef_w, bn_gamma, bn_beta,
                                   dbn_gamma, dbn_beta, gatep, coef);
  k5_final<<<dim3(NTT, N_), 256, K5_LDS_BYTES, stream>>>(x0, Wb, biasv, m_out, gatep, coef, outp);
}
